// contrastive_loss_72713796321960
// MI455X (gfx1250) — matrix operand order repaired and checked by reading, not compiled, not run on hardware
//
#include <hip/hip_runtime.h>
#include <hip/hip_bf16.h>
#include <math.h>

// ---------------------------------------------------------------------------
// NT-Xent contrastive loss, fused flash-style:
//   k1: normalize rows (f32) -> z (f16, [8192,128]) + pos logits (f32)
//   k2: per 16-row block: WMMA f16 GEMM tiles vs all 8192 cols, fixed-max
//       sum-of-exp (logits bounded by 1/T = 5); diagonal removed by exact
//       post-subtraction; sched_barrier keeps all 16 B-loads in flight
//   k3: mean(lse - pos) -> scalar loss
// ---------------------------------------------------------------------------

typedef __attribute__((ext_vector_type(16))) _Float16 v16h;
typedef __attribute__((ext_vector_type(8)))  _Float16 v8h;
typedef __attribute__((ext_vector_type(4)))  _Float16 v4h;
typedef __attribute__((ext_vector_type(8)))  float    v8f;

#define NROWS 8192      // 2B
#define BHALF 4096      // B
#define DDIM  128
#define INV_T 5.0f      // 1 / 0.2  (fixed logsumexp max: |logit| <= 5)
#define K5L2E 7.2134752044448169f   // 5 * log2(e): exp(5x-5) = exp2(K*x - K)

#if __has_builtin(__builtin_amdgcn_exp2f)
#define EXP2(x) __builtin_amdgcn_exp2f(x)
#else
#define EXP2(x) exp2f(x)
#endif

#if __has_builtin(__builtin_amdgcn_sched_barrier)
#define SCHED_FENCE() __builtin_amdgcn_sched_barrier(0)
#else
#define SCHED_FENCE()
#endif

__device__ inline float wave_sum(float v) {
#pragma unroll
    for (int m = 16; m >= 1; m >>= 1) v += __shfl_xor(v, m, 32);
    return v;
}

// --------------------------- kernel 1: normalize ---------------------------
__global__ __launch_bounds__(256) void nrm_kernel(const float* __restrict__ ei,
                                                  const float* __restrict__ ej,
                                                  _Float16* __restrict__ z,
                                                  float* __restrict__ pos) {
    const int wid  = blockIdx.x * (blockDim.x >> 5) + (threadIdx.x >> 5); // row pair
    const int lane = threadIdx.x & 31;
    if (wid >= BHALF) return;

    const float4 vi = *(const float4*)(ei + wid * DDIM + lane * 4);
    const float4 vj = *(const float4*)(ej + wid * DDIM + lane * 4);

    float si = vi.x * vi.x + vi.y * vi.y + vi.z * vi.z + vi.w * vi.w;
    float sj = vj.x * vj.x + vj.y * vj.y + vj.z * vj.z + vj.w * vj.w;
    si = wave_sum(si);
    sj = wave_sum(sj);
    const float ri = rsqrtf(si);
    const float rj = rsqrtf(sj);

    const float nix = vi.x * ri, niy = vi.y * ri, niz = vi.z * ri, niw = vi.w * ri;
    const float njx = vj.x * rj, njy = vj.y * rj, njz = vj.z * rj, njw = vj.w * rj;

    float dot = nix * njx + niy * njy + niz * njz + niw * njw;
    dot = wave_sum(dot);

    v4h zi = { (_Float16)nix, (_Float16)niy, (_Float16)niz, (_Float16)niw };
    v4h zj = { (_Float16)njx, (_Float16)njy, (_Float16)njz, (_Float16)njw };
    *(v4h*)(z + wid * DDIM + lane * 4)            = zi;
    *(v4h*)(z + (wid + BHALF) * DDIM + lane * 4)  = zj;

    if (lane == 0) pos[wid] = dot * INV_T;   // positive-pair logit (f32 exact)
}

// ---------------- kernel 2: WMMA GEMM tiles + fixed-max sum-exp ------------
__global__ __launch_bounds__(256) void lse_kernel(const _Float16* __restrict__ z,
                                                  float* __restrict__ lse) {
    const int rowBase = blockIdx.x * 16;
    const int wave    = threadIdx.x >> 5;
    const int lane    = threadIdx.x & 31;
    const int mIdx    = lane & 15;   // row-in-tile (A) / col-in-tile (B,C)
    const int hs      = lane >> 4;   // half-wave select

    // A fragments: 16x32 f16, lane(mIdx,hs) holds K in
    // [kc + hs*16, +16) (contiguous, matching B)
    v16h aF[4];
#pragma unroll
    for (int k = 0; k < 4; ++k) {
        const _Float16* p = z + (rowBase + mIdx) * DDIM + k * 32 + hs * 16;
        v8h lo = *(const v8h*)(p);
        v8h hi = *(const v8h*)(p + 8);
#pragma unroll
        for (int t = 0; t < 8; ++t) { aF[k][t] = lo[t]; aF[k][8 + t] = hi[t]; }
    }

    // squared norm of this lane's row (= the diagonal sim entry, f32):
    // the two hs groups hold disjoint K subsets of row rowBase+mIdx
    float dsq = 0.0f;
#pragma unroll
    for (int k = 0; k < 4; ++k)
#pragma unroll
        for (int t = 0; t < 16; ++t) {
            const float av = (float)aF[k][t];
            dsq = fmaf(av, av, dsq);
        }
    dsq += __shfl_xor(dsq, 16, 32);   // join the two K subsets

    float S[8];
#pragma unroll
    for (int v = 0; v < 8; ++v) S[v] = 0.0f;

    // each wave handles every 8th column tile; 2 tiles per iteration with ALL
    // 16 b128 loads issued (and fenced) before the 8 WMMAs, so each WMMA only
    // needs a partial loadcnt wait and the load pipe stays full
    for (int ct = wave; ct < NROWS / 16; ct += 16) {
        // B fragment: 32x16 f16, lane(n=mIdx,hs) holds K in [kc + hs*16, +16)
        const _Float16* q0 = z + (ct * 16       + mIdx) * DDIM + hs * 16;
        const _Float16* q1 = z + ((ct + 8) * 16 + mIdx) * DDIM + hs * 16;

        v16h b0[4], b1[4];
#pragma unroll
        for (int k = 0; k < 4; ++k) {
            b0[k] = *(const v16h*)(q0 + k * 32);
            b1[k] = *(const v16h*)(q1 + k * 32);
        }
        SCHED_FENCE();   // loads may not sink below: distinct regs, deep pipe

        v8f c0 = { 0.f, 0.f, 0.f, 0.f, 0.f, 0.f, 0.f, 0.f };
        v8f c1 = { 0.f, 0.f, 0.f, 0.f, 0.f, 0.f, 0.f, 0.f };
#pragma unroll
        for (int k = 0; k < 4; ++k) {
            c0 = __builtin_amdgcn_wmma_f32_16x16x32_f16(
                     false, aF[k], false, b0[k], (short)0, c0, false, false);
            c1 = __builtin_amdgcn_wmma_f32_16x16x32_f16(
                     false, aF[k], false, b1[k], (short)0, c1, false, false);
        }

        // unmasked: S += exp2(K*sim - K) = exp((sim*5) - 5); arg in [-14.5, ~0]
        // -> f32 safe. Diagonal contribution subtracted exactly in epilogue.
#pragma unroll
        for (int v = 0; v < 8; ++v) {
            const float e0 = EXP2(fmaf(c0[v], K5L2E, -K5L2E));
            const float e1 = EXP2(fmaf(c1[v], K5L2E, -K5L2E));
            S[v] += e0 + e1;
        }
    }

    // sum across the 16 lanes that share each row (xor <= 8 stays in-half)
#pragma unroll
    for (int off = 1; off <= 8; off <<= 1) {
#pragma unroll
        for (int v = 0; v < 8; ++v) S[v] += __shfl_xor(S[v], off, 32);
    }

    // sum the 8 waves through LDS
    __shared__ float sS[16][8];
    if (mIdx == 0) {
#pragma unroll
        for (int v = 0; v < 8; ++v) sS[v + 8 * hs][wave] = S[v];
    }
    __syncthreads();

    // threads 0..15 are wave 0, lane t -> their dsq is row rowBase+t's norm^2
    if (threadIdx.x < 16) {
        float St = 0.0f;
#pragma unroll
        for (int w = 0; w < 8; ++w) St += sS[threadIdx.x][w];
        St -= EXP2(fmaf(dsq, K5L2E, -K5L2E));             // remove diagonal
        lse[rowBase + threadIdx.x] = INV_T + __logf(St);  // fold max back in
    }
}

// ------------------------- kernel 3: final reduction -----------------------
__global__ __launch_bounds__(256) void loss_kernel(const float* __restrict__ lse,
                                                   const float* __restrict__ pos,
                                                   float* __restrict__ out) {
    float acc = 0.0f;
    for (int r = threadIdx.x; r < NROWS; r += 256)
        acc += lse[r] - pos[r & (BHALF - 1)];
    acc = wave_sum(acc);

    __shared__ float sAcc[8];
    if ((threadIdx.x & 31) == 0) sAcc[threadIdx.x >> 5] = acc;
    __syncthreads();
    if (threadIdx.x == 0) {
        float t = 0.0f;
#pragma unroll
        for (int w = 0; w < 8; ++w) t += sAcc[w];
        out[0] = t * (1.0f / (float)NROWS);
    }
}

// ------------------------------- launcher ----------------------------------
extern "C" void kernel_launch(void* const* d_in, const int* in_sizes, int n_in,
                              void* d_out, int out_size, void* d_ws, size_t ws_size,
                              hipStream_t stream) {
    (void)in_sizes; (void)n_in; (void)out_size; (void)ws_size;

    const float* ei = (const float*)d_in[0];
    const float* ej = (const float*)d_in[1];
    float* out = (float*)d_out;

    char* ws = (char*)d_ws;
    _Float16* z   = (_Float16*)ws;                                    // 2 MB
    float*    pos = (float*)(ws + (size_t)NROWS * DDIM * sizeof(_Float16));
    float*    lse = (float*)((char*)pos + (size_t)BHALF * sizeof(float));

    nrm_kernel <<<BHALF / 8,  256, 0, stream>>>(ei, ej, z, pos);
    lse_kernel <<<NROWS / 16, 256, 0, stream>>>(z, lse);
    loss_kernel<<<1,          256, 0, stream>>>(lse, pos, out);
}
